// LinearAttention3D_50818053046869
// MI455X (gfx1250) — matrix operand order repaired and checked by reading, not compiled, not run on hardware
//
#include <hip/hip_runtime.h>
#include <hip/hip_bf16.h>

// ---------------- problem constants ----------------
#define Bn 2
#define Cn 128
#define Nn 131072            // 32*64*64
#define NHEADS 4
#define DHEAD 32
#define NGROUPS 32
#define CPG 4                // channels per group
#define GRP_ELEMS (CPG * Nn) // 524288 elements per (b,group)

// ---------------- ws layout (float offsets) ----------------
static constexpr size_t GN1_SUM  = 0;
static constexpr size_t GN1_SQ   = 64;
static constexpr size_t GN1_MEAN = 128;
static constexpr size_t GN1_RSTD = 192;
static constexpr size_t QMAX     = 256;   // 256 rows, uint-encoded float max
static constexpr size_t QSUM     = 512;   // 256 rows
static constexpr size_t CTXOFF   = 768;   // B*H*32*32 = 8192 f32
static constexpr size_t W2B      = 8960;  // B*128*128 bf16 = 16384 f32 slots
static constexpr size_t GN2_SUM  = 25344;
static constexpr size_t GN2_SQ   = 25408;
static constexpr size_t GN2_MEAN = 25472;
static constexpr size_t GN2_RSTD = 25536;
static constexpr size_t WQKVB    = 25600;    // 384*128 bf16 = 24576 f32 slots
static constexpr size_t QKV_OFF  = 50176;    // B*384*N bf16 = 50,331,648 f32 slots
static constexpr size_t Y_OFF    = 50381824; // B*128*N f32

// ---------------- WMMA types ----------------
typedef __attribute__((ext_vector_type(16))) __bf16 bf16x16;
typedef __attribute__((ext_vector_type(8)))  __bf16 bf16x8;
typedef __attribute__((ext_vector_type(8)))  float  f32x8;

// monotonic float<->uint encoding for atomicMax on signed floats
__device__ __forceinline__ unsigned fenc(float f) {
    unsigned u = __float_as_uint(f);
    return (u & 0x80000000u) ? ~u : (u | 0x80000000u);
}
__device__ __forceinline__ float fdec(unsigned e) {
    return __uint_as_float((e & 0x80000000u) ? (e ^ 0x80000000u) : ~e);
}

// A-matrix fragment (16x32 bf16, row-major source, row = per-lane pointer).
// ISA layout: lanes 0-15 hold K=0..7,16..23 ; lanes 16-31 hold K=8..15,24..31
__device__ __forceinline__ bf16x16 load_a_frag(const __bf16* rowp, int kt, int lane) {
    int kb = kt * 32 + ((lane & 16) ? 8 : 0);
    bf16x8 lo = *(const bf16x8*)(rowp + kb);
    bf16x8 hi = *(const bf16x8*)(rowp + kb + 16);
    bf16x16 r;
#pragma unroll
    for (int i = 0; i < 8; ++i) { r[i] = lo[i]; r[i + 8] = hi[i]; }
    return r;
}

// B-matrix fragment (32x16 bf16), per-column pointer; lanes 0-15 hold K=0..7,16..23 ; lanes 16-31 hold K=8..15,24..31
__device__ __forceinline__ bf16x16 load_b_frag(const __bf16* colp, int kt, int lane) {
    int kb = kt * 32 + ((lane & 16) ? 8 : 0);
    bf16x8 lo = *(const bf16x8*)(colp + kb);
    bf16x8 hi = *(const bf16x8*)(colp + kb + 16);
    bf16x16 r;
#pragma unroll
    for (int i = 0; i < 8; ++i) { r[i] = lo[i]; r[i + 8] = hi[i]; }
    return r;
}

__device__ __forceinline__ f32x8 wmma_bf16(bf16x16 a, bf16x16 b, f32x8 c) {
    return __builtin_amdgcn_wmma_f32_16x16x32_bf16(false, a, false, b, (short)0, c,
                                                   false, false);
}

// ---------------- kernel 0: init stats scratch (all-zero; 0 == fenc minimum) ----
__global__ __launch_bounds__(256) void la_init_kernel(float* __restrict__ ws) {
    int i = blockIdx.x * 256 + threadIdx.x;
    if (i < 25600) ws[i] = 0.0f;
}

// ---------------- kernel 1: w_qkv -> bf16 ----------------
__global__ __launch_bounds__(256) void la_convert_w_kernel(const float* __restrict__ w,
                                                           float* __restrict__ ws) {
    int i = blockIdx.x * 256 + threadIdx.x;
    if (i < 384 * 128) ((__bf16*)(ws + WQKVB))[i] = (__bf16)w[i];
}

// ---------------- kernel 2: gn1 stats over x ----------------
__global__ __launch_bounds__(256) void la_gn_stats_kernel(const float* __restrict__ src,
                                                          float* __restrict__ sum,
                                                          float* __restrict__ sq) {
    __shared__ float s1[256], s2[256];
    int gidx = blockIdx.y;
    const float* p = src + (size_t)gidx * GRP_ELEMS + (size_t)blockIdx.x * 8192;
    float a = 0.f, aa = 0.f;
    for (int i = threadIdx.x; i < 8192; i += 256) {
        __builtin_prefetch(p + i + 2048, 0, 0);
        float v = p[i];
        a += v; aa += v * v;
    }
    s1[threadIdx.x] = a; s2[threadIdx.x] = aa;
    __syncthreads();
    for (int off = 128; off > 0; off >>= 1) {
        if ((int)threadIdx.x < off) {
            s1[threadIdx.x] += s1[threadIdx.x + off];
            s2[threadIdx.x] += s2[threadIdx.x + off];
        }
        __syncthreads();
    }
    if (threadIdx.x == 0) {
        atomicAdd(&sum[gidx], s1[0]);
        atomicAdd(&sq[gidx], s2[0]);
    }
}

__global__ void la_stats_fin_kernel(const float* __restrict__ sum,
                                    const float* __restrict__ sq,
                                    float* __restrict__ mean, float* __restrict__ rstd) {
    int i = threadIdx.x;
    if (i < 64) {
        float inv = 1.0f / (float)GRP_ELEMS;
        float m = sum[i] * inv;
        float v = sq[i] * inv - m * m;
        mean[i] = m;
        rstd[i] = rsqrtf(v + 1e-5f);
    }
}

// ---- kernel 3: qkv = W_qkv @ gn1(x) (WMMA, bf16 out) + fused q row-max ----
__global__ __launch_bounds__(256) void la_qkv_gemm_kernel(const float* __restrict__ x,
                                                          const float* __restrict__ gw,
                                                          const float* __restrict__ gb,
                                                          float* __restrict__ ws) {
    __shared__ __bf16 xs[128 * 136]; // [n][c], padded stride 136
    __shared__ unsigned rmax_enc[128];
    const float* mean = ws + GN1_MEAN;
    const float* rstd = ws + GN1_RSTD;
    const __bf16* wb = (const __bf16*)(ws + WQKVB);
    __bf16* qkvb = (__bf16*)(ws + QKV_OFF);
    unsigned* qmax = (unsigned*)(ws + QMAX);

    int blk = blockIdx.x;
    int b = blk / (Nn / 128);
    int n0 = (blk % (Nn / 128)) * 128;
    int tid = threadIdx.x;

    if (tid < 128) rmax_enc[tid] = 0u;
    for (int idx = tid; idx < 128 * 128; idx += 256) {
        int c = idx >> 7, n = idx & 127;
        float v = x[((size_t)(b * 128 + c)) * Nn + n0 + n];
        int g = b * 32 + (c >> 2);
        float xn = (v - mean[g]) * rstd[g] * gw[c] + gb[c];
        xs[n * 136 + c] = (__bf16)xn;
    }
    __syncthreads();

    int wave = tid >> 5, lane = tid & 31;
    int ncol = wave * 16;
    const __bf16* colp = xs + (size_t)(ncol + (lane & 15)) * 136;
    bf16x16 bfrag[4];
#pragma unroll
    for (int kt = 0; kt < 4; ++kt) bfrag[kt] = load_b_frag(colp, kt, lane);

    for (int mt = 0; mt < 24; ++mt) {
        f32x8 acc = {};
        const __bf16* arow = wb + (size_t)(mt * 16 + (lane & 15)) * 128;
#pragma unroll
        for (int kt = 0; kt < 4; ++kt)
            acc = wmma_bf16(load_a_frag(arow, kt, lane), bfrag[kt], acc);
        int col = n0 + ncol + (lane & 15);
        int rbase = mt * 16 + ((lane & 16) ? 8 : 0);
#pragma unroll
        for (int i = 0; i < 8; ++i)
            qkvb[((size_t)b * 384 + rbase + i) * Nn + col] = (__bf16)acc[i];
        if (mt < 8) { // q rows: fold row-max partials
#pragma unroll
            for (int i = 0; i < 8; ++i)
                atomicMax(&rmax_enc[rbase + i], fenc(acc[i]));
        }
    }
    __syncthreads();
    if (tid < 128) atomicMax(&qmax[b * 128 + tid], rmax_enc[tid]);
}

// ---------------- kernel 4: q row sum-exp ----------------
__global__ __launch_bounds__(256) void la_row_sumexp_kernel(float* __restrict__ ws) {
    __shared__ float sm[256];
    const __bf16* qkvb = (const __bf16*)(ws + QKV_OFF);
    int row = blockIdx.y;
    int b = row >> 7, hd = row & 127;
    float rmax = fdec(((const unsigned*)(ws + QMAX))[row]);
    const __bf16* p = qkvb + ((size_t)b * 384 + hd) * Nn + (size_t)blockIdx.x * 4096;
    float s = 0.f;
    for (int i = threadIdx.x; i < 4096; i += 256) {
        __builtin_prefetch(p + i + 1024, 0, 0);
        s += __expf((float)p[i] - rmax);
    }
    sm[threadIdx.x] = s;
    __syncthreads();
    for (int off = 128; off > 0; off >>= 1) {
        if ((int)threadIdx.x < off) sm[threadIdx.x] += sm[threadIdx.x + off];
        __syncthreads();
    }
    if (threadIdx.x == 0) atomicAdd(&ws[QSUM + row], sm[0]);
}

// ---------------- kernel 5: context = softmax_d(k) @ v^T (WMMA) ----------------
#define NT2 512
__global__ __launch_bounds__(256) void la_context_kernel(float* __restrict__ ws) {
    __shared__ __bf16 ksm[32 * 520];
    __shared__ __bf16 vsm[32 * 520];
    __shared__ float cpart[8 * 32 * 32];
    const __bf16* qkvb = (const __bf16*)(ws + QKV_OFF);
    float* ctx = ws + CTXOFF;
    int bh = blockIdx.y;
    int b = bh >> 2, h = bh & 3;
    size_t krow0 = (size_t)b * 384 + 128 + h * 32;
    size_t vrow0 = (size_t)b * 384 + 256 + h * 32;
    int n0 = blockIdx.x * NT2;
    int tid = threadIdx.x;

    for (int idx = tid; idx < 32 * NT2; idx += 256) {
        int e = idx >> 9, n = idx & 511;
        vsm[e * 520 + n] = qkvb[(vrow0 + e) * Nn + n0 + n];
    }
    for (int n = tid; n < NT2; n += 256) {
        float kv[32];
        float m = -1e30f;
#pragma unroll
        for (int d = 0; d < 32; ++d) {
            kv[d] = (float)qkvb[(krow0 + d) * Nn + n0 + n];
            m = fmaxf(m, kv[d]);
        }
        float s = 0.f;
#pragma unroll
        for (int d = 0; d < 32; ++d) { kv[d] = __expf(kv[d] - m); s += kv[d]; }
        float inv = 1.0f / s;
#pragma unroll
        for (int d = 0; d < 32; ++d) ksm[d * 520 + n] = (__bf16)(kv[d] * inv);
    }
    __syncthreads();

    int wave = tid >> 5, lane = tid & 31;
    int nb0 = wave * 64;
    f32x8 acc[2][2] = {};
    for (int kstep = 0; kstep < 2; ++kstep) {
        int nb = nb0 + kstep * 32;
#pragma unroll
        for (int dh = 0; dh < 2; ++dh) {
            const __bf16* arow = ksm + (size_t)(dh * 16 + (lane & 15)) * 520 + nb;
            bf16x16 af = load_a_frag(arow, 0, lane);
#pragma unroll
            for (int eh = 0; eh < 2; ++eh) {
                const __bf16* bcol = vsm + (size_t)(eh * 16 + (lane & 15)) * 520 + nb;
                bf16x16 bfr = load_b_frag(bcol, 0, lane);
                acc[dh][eh] = wmma_bf16(af, bfr, acc[dh][eh]);
            }
        }
    }
#pragma unroll
    for (int dh = 0; dh < 2; ++dh)
#pragma unroll
        for (int eh = 0; eh < 2; ++eh)
#pragma unroll
            for (int i = 0; i < 8; ++i) {
                int row = dh * 16 + i + ((lane & 16) ? 8 : 0);
                int col = eh * 16 + (lane & 15);
                cpart[wave * 1024 + row * 32 + col] = acc[dh][eh][i];
            }
    __syncthreads();
    for (int idx = tid; idx < 1024; idx += 256) {
        float s = 0.f;
#pragma unroll
        for (int w = 0; w < 8; ++w) s += cpart[w * 1024 + idx];
        atomicAdd(&ctx[(size_t)bh * 1024 + idx], s);
    }
}

// ---------------- kernel 6: W2 = (w_out . ctx^T) / qsum -> bf16 ----------------
__global__ __launch_bounds__(256) void la_make_w2_kernel(const float* __restrict__ w_out,
                                                         float* __restrict__ ws) {
    int idx = blockIdx.x * 256 + threadIdx.x;
    if (idx >= Bn * 128 * 128) return;
    int b = idx >> 14;
    int c = (idx >> 7) & 127;
    int hd = idx & 127;
    int h = hd >> 5, d = hd & 31;
    const float* cx = ws + CTXOFF + ((size_t)(b * 4 + h) * 32 + d) * 32;
    const float* wr = w_out + (size_t)c * 128 + h * 32;
    float s = 0.f;
#pragma unroll
    for (int e = 0; e < 32; ++e) s += wr[e] * cx[e];
    s /= ws[QSUM + b * 128 + hd];
    ((__bf16*)(ws + W2B))[idx] = (__bf16)s;
}

// ---- kernel 7: y = W2 @ exp(q-max) + b_out (WMMA) + fused gn2 partial stats ----
__global__ __launch_bounds__(256) void la_final_gemm_kernel(const float* __restrict__ b_out,
                                                            float* __restrict__ ws) {
    __shared__ __bf16 qe[128 * 136]; // [n][hd]
    __shared__ float smax[128];
    __shared__ float gsum[32], gsq[32];
    const __bf16* qkvb = (const __bf16*)(ws + QKV_OFF);
    const __bf16* w2b = (const __bf16*)(ws + W2B);
    float* y = ws + Y_OFF;
    int blk = blockIdx.x;
    int b = blk / (Nn / 128);
    int n0 = (blk % (Nn / 128)) * 128;
    int tid = threadIdx.x;

    if (tid < 128) smax[tid] = fdec(((const unsigned*)(ws + QMAX))[b * 128 + tid]);
    if (tid < 32) { gsum[tid] = 0.f; gsq[tid] = 0.f; }
    __syncthreads();
    for (int idx = tid; idx < 128 * 128; idx += 256) {
        int hd = idx >> 7, n = idx & 127;
        float v = (float)qkvb[((size_t)b * 384 + hd) * Nn + n0 + n];
        qe[n * 136 + hd] = (__bf16)__expf(v - smax[hd]);
    }
    __syncthreads();

    int wave = tid >> 5, lane = tid & 31;
    int ncol = wave * 16;
    const __bf16* colp = qe + (size_t)(ncol + (lane & 15)) * 136;
    bf16x16 bfrag[4];
#pragma unroll
    for (int kt = 0; kt < 4; ++kt) bfrag[kt] = load_b_frag(colp, kt, lane);

    const __bf16* w2base = w2b + (size_t)b * 128 * 128;
    for (int mt = 0; mt < 8; ++mt) {
        f32x8 acc = {};
        const __bf16* arow = w2base + (size_t)(mt * 16 + (lane & 15)) * 128;
#pragma unroll
        for (int kt = 0; kt < 4; ++kt)
            acc = wmma_bf16(load_a_frag(arow, kt, lane), bfrag[kt], acc);
        int col = n0 + ncol + (lane & 15);
        int rbase = mt * 16 + ((lane & 16) ? 8 : 0);
        float s0 = 0.f, q0 = 0.f, s1 = 0.f, q1 = 0.f;
#pragma unroll
        for (int i = 0; i < 8; ++i) {
            int c = rbase + i;
            float v = acc[i] + b_out[c];
            y[((size_t)b * 128 + c) * Nn + col] = v;
            if (i < 4) { s0 += v; q0 += v * v; } else { s1 += v; q1 += v * v; }
        }
        int g0 = rbase >> 2;
        atomicAdd(&gsum[g0], s0); atomicAdd(&gsq[g0], q0);
        atomicAdd(&gsum[g0 + 1], s1); atomicAdd(&gsq[g0 + 1], q1);
    }
    __syncthreads();
    if (tid < 32) {
        atomicAdd(&ws[GN2_SUM + b * 32 + tid], gsum[tid]);
        atomicAdd(&ws[GN2_SQ + b * 32 + tid], gsq[tid]);
    }
}

// ---------------- kernel 8: out = gn2(y) + x ----------------
__global__ __launch_bounds__(256) void la_final_apply_kernel(const float* __restrict__ x,
                                                             const float* __restrict__ g2w,
                                                             const float* __restrict__ g2b,
                                                             const float* __restrict__ ws,
                                                             float* __restrict__ out) {
    size_t i = (size_t)blockIdx.x * 256 + threadIdx.x;
    size_t base = i * 4;
    if (base >= (size_t)Bn * Cn * Nn) return;
    const float* y = ws + Y_OFF;
    size_t bc = base >> 17; // /Nn
    int c = (int)(bc & 127);
    int b = (int)(bc >> 7);
    int g = b * 32 + (c >> 2);
    float m = ws[GN2_MEAN + g], r = ws[GN2_RSTD + g];
    float scale = g2w[c] * r;
    float off = g2b[c] - m * scale;
    float4 yv = *(const float4*)(y + base);
    float4 xv = *(const float4*)(x + base);
    float4 ov;
    ov.x = yv.x * scale + off + xv.x;
    ov.y = yv.y * scale + off + xv.y;
    ov.z = yv.z * scale + off + xv.z;
    ov.w = yv.w * scale + off + xv.w;
    *(float4*)((float*)out + base) = ov;
}

// ---------------- launch ----------------
extern "C" void kernel_launch(void* const* d_in, const int* in_sizes, int n_in,
                              void* d_out, int out_size, void* d_ws, size_t ws_size,
                              hipStream_t stream) {
    const float* x     = (const float*)d_in[0];
    const float* gn1w  = (const float*)d_in[1];
    const float* gn1b  = (const float*)d_in[2];
    const float* w_qkv = (const float*)d_in[3];
    const float* w_out = (const float*)d_in[4];
    const float* b_out = (const float*)d_in[5];
    const float* gn2w  = (const float*)d_in[6];
    const float* gn2b  = (const float*)d_in[7];
    float* out = (float*)d_out;
    float* ws  = (float*)d_ws;

    la_init_kernel<<<100, 256, 0, stream>>>(ws);
    la_convert_w_kernel<<<192, 256, 0, stream>>>(w_qkv, ws);

    la_gn_stats_kernel<<<dim3(64, 64), 256, 0, stream>>>(x, ws + GN1_SUM, ws + GN1_SQ);
    la_stats_fin_kernel<<<1, 64, 0, stream>>>(ws + GN1_SUM, ws + GN1_SQ,
                                              ws + GN1_MEAN, ws + GN1_RSTD);

    la_qkv_gemm_kernel<<<Bn * (Nn / 128), 256, 0, stream>>>(x, gn1w, gn1b, ws);

    la_row_sumexp_kernel<<<dim3(32, 256), 256, 0, stream>>>(ws);

    la_context_kernel<<<dim3(Nn / NT2, Bn * NHEADS), 256, 0, stream>>>(ws);
    la_make_w2_kernel<<<128, 256, 0, stream>>>(w_out, ws);

    la_final_gemm_kernel<<<Bn * (Nn / 128), 256, 0, stream>>>(b_out, ws);
    la_stats_fin_kernel<<<1, 64, 0, stream>>>(ws + GN2_SUM, ws + GN2_SQ,
                                              ws + GN2_MEAN, ws + GN2_RSTD);

    la_final_apply_kernel<<<(Bn * Cn * (size_t)Nn) / 4 / 256, 256, 0, stream>>>(
        x, gn2w, gn2b, ws, out);
}
